// QuantizedLinearFast_23519240913001
// MI455X (gfx1250) — matrix operand order repaired and checked by reading, not compiled, not run on hardware
//
#include <hip/hip_runtime.h>

// ---- fixed problem shape (from reference) ----
#define M_DIM 8192   // TOKENS
#define N_DIM 4096   // OUT_FEATURES
#define K_DIM 4096   // IN_FEATURES
#define BLK   128    // quant block size (along K, row-major)

// ---- tiling ----
#define MT 128            // block tile M
#define NT 256            // block tile N
#define KT 64             // block tile K (two K=32 WMMA steps)
#define NTHREADS 512      // 16 waves: 4 (M) x 4 (N)
#define NIT (K_DIM / KT)  // 64 K-iterations

typedef __attribute__((ext_vector_type(16))) _Float16 v16h;
typedef __attribute__((ext_vector_type(8)))  _Float16 v8h;
typedef __attribute__((ext_vector_type(4)))  _Float16 v4h;
typedef __attribute__((ext_vector_type(2)))  _Float16 v2h;
typedef __attribute__((ext_vector_type(8)))  float    v8f;
typedef __attribute__((ext_vector_type(4)))  float    f32x4;
typedef __attribute__((ext_vector_type(4)))  int      i32x4;

// LDS row stride: 64 halves + 8 pad = 72 halves (144B) -> conflict-free frag loads
#define LDSS 72

// dequant 4 packed bytes (8 nibble codes) -> 8 f16 values (code-7)*scale
// magic: (0x6400 | c) is f16 value 1024+c (exact); +(-1031) -> c-7 (exact); *s
__device__ __forceinline__ v8h dequant8(i32x4 w, _Float16 sh) {
    union { v8h v; unsigned u[4]; v2h h2[4]; } o;
    const v2h s2  = {sh, sh};
    const v2h off = {(_Float16)(-1031.0f), (_Float16)(-1031.0f)};
#pragma unroll
    for (int j = 0; j < 4; ++j) {
        const unsigned b = (unsigned)w[j];
        o.u[j] = 0x64006400u | (b & 15u) | ((b << 12) & 0x000F0000u);
    }
#pragma unroll
    for (int j = 0; j < 4; ++j)
        o.h2[j] = (o.h2[j] + off) * s2;   // v_pk_add_f16 + v_pk_mul_f16
    return o.v;
}

__global__ __launch_bounds__(NTHREADS)
void qlin_wmma_f16_kernel(const float* __restrict__ x,
                          const int* __restrict__ pw,          // one packed byte per int32
                          const _Float16* __restrict__ scales, // fp16, one per 128 K-elems
                          float* __restrict__ out)
{
    // ping-pong LDS buffers
    __shared__ _Float16 As[2][MT * LDSS]; // x tile, f16          (2 x 18 KB)
    __shared__ _Float16 Bs[2][NT * LDSS]; // dequant W tile, f16  (2 x 36 KB)

    const int tid  = threadIdx.x;
    const int lane = tid & 31;
    const int wave = tid >> 5;
    const int wm   = wave & 3;   // 4 waves in M -> 32 rows each
    const int wn   = wave >> 2;  // 4 waves in N -> 64 cols each
    const int hk   = lane >> 4;  // lane half-group (K interleave select)
    const int ln   = lane & 15;  // row (A) / col (B,D) within 16x16 tile

    const int bm = blockIdx.y * MT;
    const int bn = blockIdx.x * NT;

    v8f acc[2][4];
#pragma unroll
    for (int i = 0; i < 2; ++i)
#pragma unroll
        for (int j = 0; j < 4; ++j)
            acc[i][j] = (v8f){0.f,0.f,0.f,0.f,0.f,0.f,0.f,0.f};

    // ---- per-thread staging registers for one K-tile ----
    // A: MT*KT floats = 8192 -> 4 float4 per thread (m = idx>>4, kg = idx&15)
    // B: NT rows * 8 groups-of-8-codes = 2048 -> 4 groups per thread
    f32x4    ax[4];
    i32x4    bw[4];
    _Float16 bsc[4];

    // ---------- prologue: load tile 0 ----------
#pragma unroll
    for (int i = 0; i < 4; ++i) {
        const int idx = tid + i * NTHREADS;
        const int m   = idx >> 4;
        const int kg  = idx & 15;
        ax[i] = *(const f32x4*)(x + (size_t)(bm + m) * K_DIM + kg * 4);
    }
#pragma unroll
    for (int i = 0; i < 4; ++i) {
        const int g  = tid + i * NTHREADS;
        const int n  = g >> 3;
        const int kg = g & 7;
        bw[i]  = *(const i32x4*)(pw + (size_t)(bn + n) * (K_DIM / 2) + kg * 4);
        bsc[i] = scales[(size_t)(bn + n) * (K_DIM / BLK)]; // block 0
    }
#pragma unroll
    for (int i = 0; i < 4; ++i) {
        const int idx = tid + i * NTHREADS;
        const int m   = idx >> 4;
        const int kg  = idx & 15;
        v4h h;
        h[0] = (_Float16)ax[i][0]; h[1] = (_Float16)ax[i][1];
        h[2] = (_Float16)ax[i][2]; h[3] = (_Float16)ax[i][3];
        *(v4h*)(&As[0][m * LDSS + kg * 4]) = h;
    }
#pragma unroll
    for (int i = 0; i < 4; ++i) {
        const int g  = tid + i * NTHREADS;
        const int n  = g >> 3;
        const int kg = g & 7;
        *(v8h*)(&Bs[0][n * LDSS + kg * 8]) = dequant8(bw[i], bsc[i]);
    }
    __syncthreads();

    // ---------- main loop: single barrier per iteration ----------
    for (int it = 0; it < NIT; ++it) {
        const int p     = it & 1;
        const int kkn   = (it + 1) * KT;
        const bool more = (it + 1) < NIT;

        // 1) issue next tile's global loads (latency hidden behind WMMAs)
        if (more) {
#pragma unroll
            for (int i = 0; i < 4; ++i) {
                const int idx = tid + i * NTHREADS;
                const int m   = idx >> 4;
                const int kg  = idx & 15;
                ax[i] = *(const f32x4*)(x + (size_t)(bm + m) * K_DIM + kkn + kg * 4);
            }
#pragma unroll
            for (int i = 0; i < 4; ++i) {
                const int g  = tid + i * NTHREADS;
                const int n  = g >> 3;
                const int kg = g & 7;
                bw[i]  = *(const i32x4*)(pw + (size_t)(bn + n) * (K_DIM / 2)
                                            + (kkn >> 1) + kg * 4);
                // all groups of a K=64 tile fall in one 128-wide scale block
                bsc[i] = scales[(size_t)(bn + n) * (K_DIM / BLK) + (kkn >> 7)];
            }
        }

        // 2) compute current tile from buffer p
#pragma unroll
        for (int ks = 0; ks < KT; ks += 32) {
            // A frag (16x32 f16): elems 0-7 -> K=hk*8+0..7 ; 8-15 -> K=16+hk*8+0..7
            union { v16h v; v8h h[2]; } a[2];
#pragma unroll
            for (int tm = 0; tm < 2; ++tm) {
                const _Float16* r = &As[p][(wm * 32 + tm * 16 + ln) * LDSS + ks];
                a[tm].h[0] = *(const v8h*)(r + hk * 8);
                a[tm].h[1] = *(const v8h*)(r + 16 + hk * 8);
            }
            // B frag (32x16 f16): lane = n + 16*hk, elems 0-7 -> K=hk*8+0..7 ; 8-15 -> K=16+hk*8+0..7
            union { v16h v; v8h h[2]; } b[4];
#pragma unroll
            for (int tn = 0; tn < 4; ++tn) {
                const _Float16* r = &Bs[p][(wn * 64 + tn * 16 + ln) * LDSS + ks];
                b[tn].h[0] = *(const v8h*)(r + hk * 8);
                b[tn].h[1] = *(const v8h*)(r + 16 + hk * 8);
            }
#pragma unroll
            for (int tm = 0; tm < 2; ++tm)
#pragma unroll
                for (int tn = 0; tn < 4; ++tn)
                    acc[tm][tn] = __builtin_amdgcn_wmma_f32_16x16x32_f16(
                        false, a[tm].v, false, b[tn].v,
                        (short)0, acc[tm][tn], false, false);
        }

        // 3) convert + store next tile into buffer 1-p (its readers finished
        //    before the previous barrier)
        if (more) {
            const int q = p ^ 1;
#pragma unroll
            for (int i = 0; i < 4; ++i) {
                const int idx = tid + i * NTHREADS;
                const int m   = idx >> 4;
                const int kg  = idx & 15;
                v4h h;
                h[0] = (_Float16)ax[i][0]; h[1] = (_Float16)ax[i][1];
                h[2] = (_Float16)ax[i][2]; h[3] = (_Float16)ax[i][3];
                *(v4h*)(&As[q][m * LDSS + kg * 4]) = h;
            }
#pragma unroll
            for (int i = 0; i < 4; ++i) {
                const int g  = tid + i * NTHREADS;
                const int n  = g >> 3;
                const int kg = g & 7;
                *(v8h*)(&Bs[q][n * LDSS + kg * 8]) = dequant8(bw[i], bsc[i]);
            }
        }
        __syncthreads();
    }

    // ---- epilogue: D layout VGPR v -> M = tm*16 + hk*8 + v, N = ln ----
    // Non-temporal stores: keep x/W resident in the 192MB L2, stream the output.
#pragma unroll
    for (int tm = 0; tm < 2; ++tm)
#pragma unroll
        for (int tn = 0; tn < 4; ++tn) {
            const int n = bn + wn * 64 + tn * 16 + ln;
            float* op = out + (size_t)(bm + wm * 32 + tm * 16 + hk * 8) * N_DIM + n;
#pragma unroll
            for (int v = 0; v < 8; ++v)
                __builtin_nontemporal_store(acc[tm][tn][v], op + (size_t)v * N_DIM);
        }
}

extern "C" void kernel_launch(void* const* d_in, const int* in_sizes, int n_in,
                              void* d_out, int out_size, void* d_ws, size_t ws_size,
                              hipStream_t stream) {
    (void)in_sizes; (void)n_in; (void)d_ws; (void)ws_size; (void)out_size;
    const float*    x  = (const float*)d_in[0];
    const int*      pw = (const int*)d_in[1];
    const _Float16* sc = (const _Float16*)d_in[2];
    float*          y  = (float*)d_out;

    dim3 grid(N_DIM / NT, M_DIM / MT); // (16, 64)
    qlin_wmma_f16_kernel<<<grid, NTHREADS, 0, stream>>>(x, pw, sc, y);
}
